// QuantizedLinear_19550691131799
// MI455X (gfx1250) — compile-verified, matrix operand order checked by reading, not run on hardware
//
#include <hip/hip_runtime.h>

// ---------------------------------------------------------------------------
// QuantizedLinear for MI455X (gfx1250, wave32, WMMA)
//   out[B,O] = x[B,I] @ W[O,I]^T,  W = dequant(int4,scales) + U@V^T, rows
//   overwritten by protected channels.  B=O=I=4096, GROUP=128, RANK=32.
// Strategy: materialize W and x in f16 (both fit in 192MB L2), then a
// register-blocked v_wmma_f32_16x16x32_f16 GEMM, f32 accumulate.
// Wave tile 64x64: 16 WMMAs per K-step from 16 b128 loads (16 B/lane/WMMA).
// ---------------------------------------------------------------------------

typedef __attribute__((ext_vector_type(16))) _Float16 v16h;
typedef __attribute__((ext_vector_type(4)))  _Float16 v4h;
typedef __attribute__((ext_vector_type(8)))  float    v8f;

#define OUT_F 4096
#define IN_F  4096
#define GROUPSZ 128
#define RANK  32
#define NPROT 32
#define BATCH 4096

// ---------------------------------------------------------------- x -> f16
__global__ void ql_convert_x_f16(const float* __restrict__ x,
                                 _Float16* __restrict__ xh, int n4) {
  int i = blockIdx.x * blockDim.x + threadIdx.x;
  if (i < n4) {
    float4 v = reinterpret_cast<const float4*>(x)[i];
    v4h h;
    h[0] = (_Float16)v.x; h[1] = (_Float16)v.y;
    h[2] = (_Float16)v.z; h[3] = (_Float16)v.w;
    reinterpret_cast<v4h*>(xh)[i] = h;
  }
}

// ------------------------------------------- dequant + low-rank correction
// Split-nibble layout: flat f = o*IN + i; f < OUT*IN/2 -> low nibble of
// packed[f]; else high nibble of packed[f - OUT*IN/2].
__global__ void ql_build_w(const int* __restrict__ packed,
                           const float* __restrict__ scales,
                           const float* __restrict__ svd_u,
                           const float* __restrict__ svd_v,
                           _Float16* __restrict__ wh) {
  __shared__ float su[RANK];
  const int o = blockIdx.x;
  if (threadIdx.x < RANK) su[threadIdx.x] = svd_u[o * RANK + threadIdx.x];
  __syncthreads();

  const int HALF = (OUT_F * IN_F) / 2;
  #pragma unroll 2
  for (int c = 0; c < IN_F / 256; ++c) {
    int i = c * 256 + threadIdx.x;
    int f = o * IN_F + i;
    int q = (f < HALF) ? (packed[f] & 15) : ((packed[f - HALF] >> 4) & 15);
    float val = (float)(q - 8) * scales[i >> 7];   // GROUPSZ = 128
    const float4* vp = reinterpret_cast<const float4*>(svd_v + (size_t)i * RANK);
    float acc = 0.0f;
    #pragma unroll
    for (int r = 0; r < RANK / 4; ++r) {
      float4 vv = vp[r];
      acc += vv.x * su[4 * r + 0] + vv.y * su[4 * r + 1] +
             vv.z * su[4 * r + 2] + vv.w * su[4 * r + 3];
    }
    wh[f] = (_Float16)(val + acc);
  }
}

// ---------------------------------------------------- protected-row rewrite
__global__ void ql_protect_rows(const float* __restrict__ prot,
                                const int* __restrict__ pidx,
                                _Float16* __restrict__ wh) {
  int p = blockIdx.x;
  int o = pidx[p];
  for (int c = threadIdx.x; c < IN_F; c += blockDim.x)
    wh[(size_t)o * IN_F + c] = (_Float16)prot[(size_t)p * IN_F + c];
}

// ----------------------------------------------------------- WMMA fragment
// 16-bit A 16x32 layout (ISA 7.12.2): lanes 0-15 hold K 0..7 & 16..23,
// lanes 16-31 hold K 8..15 & 24..31 -> two contiguous b128 loads per lane.
// B (=W^T tile) is the mirror: lane n holds W-row n with the same K pattern,
// so the identical loader serves both operands.
__device__ __forceinline__ v16h ql_load_frag(const _Float16* p) {
  union { float4 f[2]; v16h v; } u;
  u.f[0] = *reinterpret_cast<const float4*>(p);
  u.f[1] = *reinterpret_cast<const float4*>(p + 16);
  return u.v;
}

// -------------------------------------------------------------- WMMA GEMM
// 256 threads = 8 waves, arranged 2(M) x 4(N). Wave tile 64x64 (16 wmma per
// K-step from 16 b128 loads), workgroup tile 128x256.
// Grid: (OUT/256, BATCH/128).
__global__ void __launch_bounds__(256)
ql_wmma_gemm(const _Float16* __restrict__ xh,
             const _Float16* __restrict__ wh,
             float* __restrict__ out) {
  const int lane = threadIdx.x & 31;
  const int wave = threadIdx.x >> 5;
  const int wm = wave >> 2;            // 0..1
  const int wn = wave & 3;             // 0..3
  const int l15 = lane & 15;
  const int half = lane >> 4;
  const int koff = half * 8;           // per-lane K sub-offset (elements)

  const int m0 = blockIdx.y * 128 + wm * 64;  // batch rows of this wave
  const int n0 = blockIdx.x * 256 + wn * 64;  // output cols of this wave

  v8f acc[4][4];
  #pragma unroll
  for (int i = 0; i < 4; ++i)
    #pragma unroll
    for (int j = 0; j < 4; ++j)
      acc[i][j] = {};

  const _Float16* xrow[4];
  const _Float16* wrow[4];
  #pragma unroll
  for (int i = 0; i < 4; ++i)
    xrow[i] = xh + (size_t)(m0 + 16 * i + l15) * IN_F + koff;
  #pragma unroll
  for (int j = 0; j < 4; ++j)
    wrow[j] = wh + (size_t)(n0 + 16 * j + l15) * IN_F + koff;

  for (int k = 0; k < IN_F; k += 32) {
    v16h a[4], b[4];
    #pragma unroll
    for (int i = 0; i < 4; ++i) a[i] = ql_load_frag(xrow[i] + k);
    #pragma unroll
    for (int j = 0; j < 4; ++j) b[j] = ql_load_frag(wrow[j] + k);

    #pragma unroll
    for (int i = 0; i < 4; ++i)
      #pragma unroll
      for (int j = 0; j < 4; ++j)
        acc[i][j] = __builtin_amdgcn_wmma_f32_16x16x32_f16(
            false, a[i], false, b[j], (short)0, acc[i][j], false, false);
  }

  // C/D layout: VGPR r -> row (half*8 + r), lane l15 -> col.
  #pragma unroll
  for (int i = 0; i < 4; ++i) {
    int mrow = m0 + i * 16 + half * 8;
    #pragma unroll
    for (int j = 0; j < 4; ++j) {
      int col = n0 + j * 16 + l15;
      #pragma unroll
      for (int r = 0; r < 8; ++r)
        out[(size_t)(mrow + r) * OUT_F + col] = acc[i][j][r];
    }
  }
}

// ---------------------------------------------------------------------------
extern "C" void kernel_launch(void* const* d_in, const int* in_sizes, int n_in,
                              void* d_out, int out_size, void* d_ws, size_t ws_size,
                              hipStream_t stream) {
  const float* x      = (const float*)d_in[0];
  const int*   packed = (const int*)d_in[1];
  const float* scales = (const float*)d_in[2];
  const float* svd_u  = (const float*)d_in[3];
  const float* svd_v  = (const float*)d_in[4];
  const float* prot   = (const float*)d_in[5];
  const int*   pidx   = (const int*)d_in[6];
  float* out = (float*)d_out;

  // Workspace: [0,32MB) W in f16, [32MB,64MB) x in f16.
  _Float16* wh = (_Float16*)d_ws;
  _Float16* xh = (_Float16*)((char*)d_ws + (size_t)OUT_F * IN_F * sizeof(_Float16));

  ql_convert_x_f16<<<(BATCH * IN_F / 4 + 255) / 256, 256, 0, stream>>>(
      x, xh, BATCH * IN_F / 4);
  ql_build_w<<<OUT_F, 256, 0, stream>>>(packed, scales, svd_u, svd_v, wh);
  ql_protect_rows<<<NPROT, 256, 0, stream>>>(prot, pidx, wh);
  ql_wmma_gemm<<<dim3(OUT_F / 256, BATCH / 128), 256, 0, stream>>>(xh, wh, out);
}
